// MoELLMMini_50422916055542
// MI455X (gfx1250) — matrix operand order repaired and checked by reading, not compiled, not run on hardware
//
#include <hip/hip_runtime.h>

typedef unsigned short u16;
typedef __attribute__((ext_vector_type(16))) __bf16 v16bf;
typedef __attribute__((ext_vector_type(8)))  __bf16 v8bf;
typedef __attribute__((ext_vector_type(8)))  float  v8f;
typedef __attribute__((ext_vector_type(4)))  int    v4i;

#if defined(__gfx1250__) && __has_builtin(__builtin_amdgcn_global_load_async_to_lds_b128)
#define ASYNC_LDS 1
#else
#define ASYNC_LDS 0
#endif

#define GAS __attribute__((address_space(1)))
#define LAS __attribute__((address_space(3)))

// ---------------- bf16 helpers ----------------
static __device__ __forceinline__ __bf16 f2bf(float f){
  unsigned u = __builtin_bit_cast(unsigned, f);
  u += 0x7fffu + ((u >> 16) & 1u);                 // round-to-nearest-even
  return __builtin_bit_cast(__bf16, (unsigned short)(u >> 16));
}
static __device__ __forceinline__ u16 f2bfu(float f){
  return __builtin_bit_cast(u16, f2bf(f));
}

// ---------------- WMMA wrapper ----------------
static __device__ __forceinline__ v8f wmma_bf16(v16bf a, v16bf b, v8f c){
  // (neg_a, A, neg_b, B, c_mod, C, reuse_a, reuse_b)
  return __builtin_amdgcn_wmma_f32_16x16x32_bf16(false, a, false, b, (short)0, c,
                                                 false, false);
}

// ---------------- async / wait helpers ----------------
static __device__ __forceinline__ void async_cp16(const u16* gp, u16* lp){
#if ASYNC_LDS
  // signature (probe-confirmed by diagnostic): (v4i32 AS1*, v4i32 AS3*, imm offset, imm cpol)
  __builtin_amdgcn_global_load_async_to_lds_b128((GAS v4i*)(void*)gp,
                                                 (LAS v4i*)(void*)lp, 0, 0);
#else
  *(uint4*)lp = *(const uint4*)gp;   // sync fallback: global_load_b128 + ds_store_b128
#endif
}
static __device__ __forceinline__ void wait_async_le4(){
#if ASYNC_LDS
#if __has_builtin(__builtin_amdgcn_s_wait_asynccnt)
  __builtin_amdgcn_s_wait_asynccnt(4);
#else
  asm volatile("s_wait_asynccnt 0x4" ::: "memory");
#endif
#endif
}
static __device__ __forceinline__ void wait_async_le0(){
#if ASYNC_LDS
#if __has_builtin(__builtin_amdgcn_s_wait_asynccnt)
  __builtin_amdgcn_s_wait_asynccnt(0);
#else
  asm volatile("s_wait_asynccnt 0x0" ::: "memory");
#endif
#endif
}

// ---------------- fragment loads ----------------
// A-matrix 16x32 bf16 fragment from global [*,lda]:
// lanes 0-15: K = 0..15; lanes 16-31: K = 16..31; M = lane&15
static __device__ __forceinline__ v16bf load_afrag(const u16* __restrict__ A, int lda,
                                                   int m0, int k0, int lane){
  int l = lane & 15, hi = lane >> 4;
  const u16* p = A + (size_t)(m0 + l) * lda + k0 + hi * 16;
  v8bf x0 = *(const v8bf*)p;
  v8bf x1 = *(const v8bf*)(p + 8);
  v16bf r;
#pragma unroll
  for (int i = 0; i < 8; i++){ r[i] = x0[i]; r[8 + i] = x1[i]; }
  return r;
}
// B-matrix 32x16 bf16 fragment from [N,K]-row-major global weight
static __device__ __forceinline__ v16bf load_bfrag(const u16* __restrict__ W, int ldw,
                                                   int n0, int k0, int lane){
  int l = lane & 15, hi = lane >> 4;
  const u16* p = W + (size_t)(n0 + l) * ldw + k0 + hi * 16;
  return *(const v16bf*)p;
}

// LDS tile: 128 rows x 32 K, padded row stride 40 u16 (80B -> conflict-free b128 reads)
#define LROW 40
static __device__ __forceinline__ v16bf lds_afrag(const u16* s, int m0, int lane){
  int l = lane & 15, hi = lane >> 4;
  const u16* p = s + (m0 + l) * LROW + hi * 8;
  v8bf x0 = *(const v8bf*)p;
  v8bf x1 = *(const v8bf*)(p + 16);
  v16bf r;
#pragma unroll
  for (int i = 0; i < 8; i++){ r[i] = x0[i]; r[8 + i] = x1[i]; }
  return r;
}
static __device__ __forceinline__ v16bf lds_bfrag(const u16* s, int n0, int lane){
  int l = lane & 15, hi = lane >> 4;
  const u16* p = s + (n0 + l) * LROW + hi * 8;
  v8bf x0 = *(const v8bf*)p;
  v8bf x1 = *(const v8bf*)(p + 16);
  v16bf r;
#pragma unroll
  for (int i = 0; i < 8; i++){ r[i] = x0[i]; r[8 + i] = x1[i]; }
  return r;
}

// stage a 128-row x 32-K bf16 tile into LDS (2 x 16B chunks per thread,
// i.e. 4 async-to-LDS instructions per wave per (A,B) tile pair)
static __device__ __forceinline__ void stage_tile(const u16* __restrict__ g, int ld,
                                                  int row0, int k0, u16* s, int tid){
#pragma unroll
  for (int i = 0; i < 2; i++){
    int c = tid + i * 256;           // chunk id 0..511
    int r = c >> 2;                  // row 0..127
    int o = (c & 3) * 8;             // u16 offset within row (0,8,16,24)
    async_cp16(g + (size_t)(row0 + r) * ld + k0 + o, s + r * LROW + o);
  }
}

// ---------------- bf16 WMMA GEMM, LDS double-buffered ----------------
// C[M,N] (fp32) = op( A_bf16[M,K] @ W_bf16[N,K]^T + bias )
// block = 256 thr (8 waves); block tile 128x128; wave tile 32Mx64N (8 WMMA / k-step)
__global__ __launch_bounds__(256) void gemm_kernel(
    const u16* __restrict__ A, int lda,
    const u16* __restrict__ W, int ldw,
    const float* __restrict__ bias,
    const float* __restrict__ rowscale, int rs_stride,
    float* __restrict__ C, int ldc, u16* __restrict__ Cb,
    int K, int relu, int accum)
{
  __shared__ u16 sA[2][128 * LROW];
  __shared__ u16 sB[2][128 * LROW];
  int tid  = threadIdx.x;
  int lane = tid & 31;
  int wave = tid >> 5;
  int mblk = blockIdx.y * 128, nblk = blockIdx.x * 128;
  int mw = (wave & 3) * 32;          // wave M offset inside block tile
  int nw = (wave >> 2) * 64;         // wave N offset inside block tile

  v8f acc[2][4];
#pragma unroll
  for (int mi = 0; mi < 2; mi++)
#pragma unroll
    for (int ni = 0; ni < 4; ni++)
#pragma unroll
      for (int r = 0; r < 8; r++) acc[mi][ni][r] = 0.f;

  stage_tile(A, lda, mblk, 0, sA[0], tid);
  stage_tile(W, ldw, nblk, 0, sB[0], tid);

  int nk = K >> 5;
  for (int ki = 0; ki < nk; ki++){
    int cur = ki & 1;
    if (ki + 1 < nk){
      stage_tile(A, lda, mblk, (ki + 1) * 32, sA[cur ^ 1], tid);
      stage_tile(W, ldw, nblk, (ki + 1) * 32, sB[cur ^ 1], tid);
      wait_async_le4();              // own prev-tile issues complete (in-order)
    } else {
      wait_async_le0();
    }
    __syncthreads();                 // all waves waited -> tile visible block-wide

    v16bf af0 = lds_afrag(sA[cur], mw,      lane);
    v16bf af1 = lds_afrag(sA[cur], mw + 16, lane);
    v16bf bf0 = lds_bfrag(sB[cur], nw,      lane);
    v16bf bf1 = lds_bfrag(sB[cur], nw + 16, lane);
    v16bf bf2 = lds_bfrag(sB[cur], nw + 32, lane);
    v16bf bf3 = lds_bfrag(sB[cur], nw + 48, lane);
    acc[0][0] = wmma_bf16(af0, bf0, acc[0][0]);
    acc[0][1] = wmma_bf16(af0, bf1, acc[0][1]);
    acc[0][2] = wmma_bf16(af0, bf2, acc[0][2]);
    acc[0][3] = wmma_bf16(af0, bf3, acc[0][3]);
    acc[1][0] = wmma_bf16(af1, bf0, acc[1][0]);
    acc[1][1] = wmma_bf16(af1, bf1, acc[1][1]);
    acc[1][2] = wmma_bf16(af1, bf2, acc[1][2]);
    acc[1][3] = wmma_bf16(af1, bf3, acc[1][3]);

    __syncthreads();                 // done reading before this buffer is re-staged
  }

  int l = lane & 15, hi = lane >> 4;
#pragma unroll
  for (int mi = 0; mi < 2; mi++){
#pragma unroll
    for (int r = 0; r < 8; r++){
      int row = mblk + mw + mi * 16 + hi * 8 + r;
      float rs = rowscale ? rowscale[(size_t)row * rs_stride] : 1.f;
#pragma unroll
      for (int ni = 0; ni < 4; ni++){
        int col = nblk + nw + ni * 16 + l;
        float v = acc[mi][ni][r];
        if (bias)     v += bias[col];
        if (relu)     v  = fmaxf(v, 0.f);
        if (rowscale) v *= rs;
        size_t idx = (size_t)row * ldc + col;
        if (accum) C[idx] += v; else C[idx] = v;
        if (Cb) Cb[idx] = f2bfu(v);
      }
    }
  }
}

// ---------------- flash attention ----------------
// qkv bf16 [T,1536] (q|k|v each D=512, 8 heads x 64). One wave = 16 q rows.
// Scores computed transposed (K x Q^T) so exp(P) repacks in-lane into an A-fragment.
__global__ __launch_bounds__(256) void attn_kernel(const u16* __restrict__ qkv,
                                                   u16* __restrict__ ob)
{
  int lane = threadIdx.x & 31;
  int wave = threadIdx.x >> 5;
  int b = blockIdx.z, h = blockIdx.y;
  int q0 = blockIdx.x * 128 + wave * 16;
  const u16* qb = qkv + (size_t)b * 2048 * 1536 + h * 64;
  const u16* kb = qb + 512;
  const u16* vb = qb + 1024;

  v16bf bq0 = load_bfrag(qb, 1536, q0, 0, lane);   // B-frag: n=q, k=d
  v16bf bq1 = load_bfrag(qb, 1536, q0, 32, lane);

  int l = lane & 15, hi = lane >> 4;
  float m_i = -3.0e38f, l_i = 0.f;                 // stats for q = q0 + (lane&15)
  v8f oacc[4];
#pragma unroll
  for (int d = 0; d < 4; d++)
#pragma unroll
    for (int r = 0; r < 8; r++) oacc[d][r] = 0.f;

  for (int kt = 0; kt < 2048; kt += 32){
    v16bf ak0 = load_afrag(kb, 1536, kt,      0,  lane);
    v16bf ak1 = load_afrag(kb, 1536, kt,      32, lane);
    v8f s0; v8f s1;
#pragma unroll
    for (int r = 0; r < 8; r++){ s0[r] = 0.f; s1[r] = 0.f; }
    s0 = wmma_bf16(ak0, bq0, s0);
    s0 = wmma_bf16(ak1, bq1, s0);
    v16bf ak2 = load_afrag(kb, 1536, kt + 16, 0,  lane);
    v16bf ak3 = load_afrag(kb, 1536, kt + 16, 32, lane);
    s1 = wmma_bf16(ak2, bq0, s1);
    s1 = wmma_bf16(ak3, bq1, s1);

    float mx = -3.0e38f;
#pragma unroll
    for (int r = 0; r < 8; r++){
      float a = s0[r] * 0.125f, c = s1[r] * 0.125f;  // 1/sqrt(64)
      s0[r] = a; s1[r] = c;
      mx = fmaxf(mx, fmaxf(a, c));
    }
    mx = fmaxf(mx, __shfl_xor(mx, 16, 32));          // other kv half, same q
    float mnew  = fmaxf(m_i, mx);
    float alpha = __expf(m_i - mnew);
    float psum  = 0.f;
    v16bf ap;                                        // P as A-frag (M=q, K=kv)
#pragma unroll
    for (int r = 0; r < 8; r++){
      float p0 = __expf(s0[r] - mnew);
      float p1 = __expf(s1[r] - mnew);
      psum += p0 + p1;
      ap[r] = f2bf(p0); ap[8 + r] = f2bf(p1);
    }
    psum += __shfl_xor(psum, 16, 32);
    l_i = l_i * alpha + psum;
    m_i = mnew;
#pragma unroll
    for (int r = 0; r < 8; r++){
      float ar = __shfl(alpha, hi * 8 + r, 32);      // alpha of O-row q = hi*8+r
      oacc[0][r] *= ar; oacc[1][r] *= ar; oacc[2][r] *= ar; oacc[3][r] *= ar;
    }
#pragma unroll
    for (int dt = 0; dt < 4; dt++){
      const u16* vp = vb + (size_t)(kt + hi * 8) * 1536 + dt * 16 + l;
      v16bf bv;
#pragma unroll
      for (int i = 0; i < 16; i++)
        bv[i] = __builtin_bit_cast(__bf16, vp[(size_t)(i + (i & 8)) * 1536]);
      oacc[dt] = wmma_bf16(ap, bv, oacc[dt]);
    }
  }
#pragma unroll
  for (int r = 0; r < 8; r++){
    float lr  = __shfl(l_i, hi * 8 + r, 32);
    float inv = 1.0f / lr;
    size_t row = (size_t)b * 2048 + q0 + hi * 8 + r;
#pragma unroll
    for (int dt = 0; dt < 4; dt++)
      ob[row * 512 + h * 64 + dt * 16 + l] = f2bfu(oacc[dt][r] * inv);
  }
}

// ---------------- add + LayerNorm (D=512, block=256, 2 elems/thr) ----------------
__global__ __launch_bounds__(256) void add_ln_kernel(
    const float* __restrict__ x, const float* __restrict__ a,
    const float* __restrict__ g, const float* __restrict__ b,
    float* __restrict__ y, u16* __restrict__ yb)
{
  __shared__ float red[256];
  int tid = threadIdx.x;
  size_t base = (size_t)blockIdx.x * 512;
  float v0 = x[base + tid], v1 = x[base + 256 + tid];
  if (a){ v0 += a[base + tid]; v1 += a[base + 256 + tid]; }
  red[tid] = v0 + v1; __syncthreads();
  for (int o = 128; o > 0; o >>= 1){ if (tid < o) red[tid] += red[tid + o]; __syncthreads(); }
  float mean = red[0] * (1.f / 512.f);
  __syncthreads();
  float d0 = v0 - mean, d1 = v1 - mean;
  red[tid] = d0 * d0 + d1 * d1; __syncthreads();
  for (int o = 128; o > 0; o >>= 1){ if (tid < o) red[tid] += red[tid + o]; __syncthreads(); }
  float inv = rsqrtf(red[0] * (1.f / 512.f) + 1e-5f);
  float r0 = d0 * inv * g[tid]       + b[tid];
  float r1 = d1 * inv * g[256 + tid] + b[256 + tid];
  y[base + tid] = r0;        y[base + 256 + tid] = r1;
  yb[base + tid] = f2bfu(r0); yb[base + 256 + tid] = f2bfu(r1);
}

// ---------------- embedding gather ----------------
__global__ __launch_bounds__(256) void embed_kernel(
    const int* __restrict__ ids, const float* __restrict__ emb,
    float* __restrict__ x, u16* __restrict__ xb)
{
  int tid = threadIdx.x;
  size_t src = (size_t)ids[blockIdx.x] * 512;
  size_t dst = (size_t)blockIdx.x * 512;
  float v0 = emb[src + tid], v1 = emb[src + 256 + tid];
  x[dst + tid] = v0;         x[dst + 256 + tid] = v1;
  xb[dst + tid] = f2bfu(v0); xb[dst + 256 + tid] = f2bfu(v1);
}

// ---------------- weight convert / transpose ----------------
__global__ __launch_bounds__(256) void cvt_kernel(const float* __restrict__ in,
                                                  u16* __restrict__ out, size_t n)
{
  size_t i = (size_t)blockIdx.x * 256 + threadIdx.x;
  if (i < n) out[i] = f2bfu(in[i]);
}
// in fp32 [R,C] -> out bf16 [C,R]
__global__ __launch_bounds__(256) void cvtT_kernel(const float* __restrict__ in,
                                                   u16* __restrict__ out, int R, int C)
{
  size_t i = (size_t)blockIdx.x * 256 + threadIdx.x;
  if (i >= (size_t)R * C) return;
  int r = (int)(i / C), c = (int)(i % C);
  out[(size_t)c * R + r] = f2bfu(in[i]);
}

__global__ __launch_bounds__(256) void zero_kernel(float* __restrict__ p, size_t n)
{
  size_t i = (size_t)blockIdx.x * 256 + threadIdx.x;
  if (i < n) p[i] = 0.f;
}

// ---------------- MoE gating ----------------
__global__ __launch_bounds__(256) void gate_kernel(
    const float* __restrict__ x, const float* __restrict__ gw,
    const float* __restrict__ gb, float* __restrict__ gs)
{
  int idx = blockIdx.x * 256 + threadIdx.x;
  if (idx >= 4096 * 8) return;
  int t = idx >> 3, e = idx & 7;
  const float* xr = x + (size_t)t * 512;
  const float* wr = gw + (size_t)e * 512;
  float acc = gb[e];
  for (int k = 0; k < 512; k++) acc += xr[k] * wr[k];
  gs[idx] = acc;
}

__global__ __launch_bounds__(256) void topk_kernel(const float* __restrict__ gs,
                                                   float* __restrict__ we)
{
  int t = blockIdx.x * 256 + threadIdx.x;
  if (t >= 4096) return;
  float s[8];
#pragma unroll
  for (int e = 0; e < 8; e++) s[e] = gs[(size_t)t * 8 + e];
  int i0 = 0;
#pragma unroll
  for (int e = 1; e < 8; e++) if (s[e] > s[i0]) i0 = e;
  int i1 = (i0 == 0) ? 1 : 0;
#pragma unroll
  for (int e = 0; e < 8; e++) if (e != i0 && s[e] > s[i1]) i1 = e;
  float m  = fmaxf(s[i0], s[i1]);
  float e0 = __expf(s[i0] - m), e1 = __expf(s[i1] - m);
  float inv = 1.f / (e0 + e1);
#pragma unroll
  for (int e = 0; e < 8; e++) we[(size_t)t * 8 + e] = 0.f;
  we[(size_t)t * 8 + i0] = e0 * inv;
  we[(size_t)t * 8 + i1] = e1 * inv;
}

// ---------------- host orchestration ----------------
extern "C" void kernel_launch(void* const* d_in, const int* in_sizes, int n_in,
                              void* d_out, int out_size, void* d_ws, size_t ws_size,
                              hipStream_t stream)
{
  (void)in_sizes; (void)n_in; (void)out_size; (void)ws_size;
  const int B = 2, S = 2048, D = 512, F = 2048, E = 8, V = 32000, H = 8;
  const int T = B * S; // 4096

  const int*   ids    = (const int*)d_in[0];
  const float* emb    = (const float*)d_in[1];
  const float* lnf_g  = (const float*)d_in[30];
  const float* lnf_b  = (const float*)d_in[31];
  const float* head_w = (const float*)d_in[32];
  const float* head_b = (const float*)d_in[33];

  char* wsp = (char*)d_ws;
  size_t off = 0;
  auto alloc = [&](size_t bytes) -> void* {
    void* r = wsp + off;
    off += (bytes + 255) & ~(size_t)255;
    return r;
  };
  u16*   xb    = (u16*)  alloc((size_t)T * D * 2);
  float* x     = (float*)alloc((size_t)T * D * 4);
  float* y     = (float*)alloc((size_t)T * D * 4);
  u16*   yb    = (u16*)  alloc((size_t)T * D * 2);
  float* qkvf  = (float*)alloc((size_t)T * 3 * D * 4);
  u16*   qkvb  = (u16*)  alloc((size_t)T * 3 * D * 2);
  u16*   ob    = (u16*)  alloc((size_t)T * D * 2);
  float* abuf  = (float*)alloc((size_t)T * D * 4);
  float* gs    = (float*)alloc((size_t)T * E * 4);
  float* we    = (float*)alloc((size_t)T * E * 4);
  float* hf    = (float*)alloc((size_t)T * F * 4);
  u16*   hb    = (u16*)  alloc((size_t)T * F * 2);
  float* macc  = (float*)alloc((size_t)T * D * 4);
  u16*   inw_b = (u16*)  alloc((size_t)3 * D * D * 2);
  u16*   outw_b= (u16*)  alloc((size_t)D * D * 2);
  u16*   w1t   = (u16*)  alloc((size_t)E * D * F * 2);
  u16*   w2t   = (u16*)  alloc((size_t)E * F * D * 2);
  u16*   hwb   = (u16*)  alloc((size_t)V * D * 2);

  embed_kernel<<<T, 256, 0, stream>>>(ids, emb, x, xb);
  cvt_kernel<<<(unsigned)(((size_t)V * D + 255) / 256), 256, 0, stream>>>(head_w, hwb, (size_t)V * D);

  float* curx = x;  u16* curxb = xb;
  float* othx = y;  u16* othxb = yb;

  for (int L = 0; L < 2; L++){
    void* const* lp = d_in + 2 + L * 14;
    const float* in_w  = (const float*)lp[0];
    const float* in_b  = (const float*)lp[1];
    const float* out_w = (const float*)lp[2];
    const float* out_b = (const float*)lp[3];
    const float* n1g   = (const float*)lp[4];
    const float* n1b   = (const float*)lp[5];
    const float* n2g   = (const float*)lp[6];
    const float* n2b   = (const float*)lp[7];
    const float* gw    = (const float*)lp[8];
    const float* gb    = (const float*)lp[9];
    const float* w1    = (const float*)lp[10];
    const float* b1    = (const float*)lp[11];
    const float* w2    = (const float*)lp[12];
    const float* b2    = (const float*)lp[13];

    cvt_kernel<<<(3 * D * D + 255) / 256, 256, 0, stream>>>(in_w, inw_b, (size_t)3 * D * D);
    cvt_kernel<<<(D * D + 255) / 256, 256, 0, stream>>>(out_w, outw_b, (size_t)D * D);
    for (int e = 0; e < E; e++){
      cvtT_kernel<<<(D * F + 255) / 256, 256, 0, stream>>>(w1 + (size_t)e * D * F,
                                                           w1t + (size_t)e * F * D, D, F);
      cvtT_kernel<<<(F * D + 255) / 256, 256, 0, stream>>>(w2 + (size_t)e * F * D,
                                                           w2t + (size_t)e * D * F, F, D);
    }

    // qkv = xb @ in_w^T + in_b  (also bf16 shadow)
    gemm_kernel<<<dim3(3 * D / 128, T / 128), 256, 0, stream>>>(
        curxb, D, inw_b, D, in_b, nullptr, 0, qkvf, 3 * D, qkvb, D, 0, 0);
    attn_kernel<<<dim3(S / 128, H, B), 256, 0, stream>>>(qkvb, ob);
    // a = attn_out @ out_w^T + out_b
    gemm_kernel<<<dim3(D / 128, T / 128), 256, 0, stream>>>(
        ob, D, outw_b, D, out_b, nullptr, 0, abuf, D, nullptr, D, 0, 0);
    add_ln_kernel<<<T, 256, 0, stream>>>(curx, abuf, n1g, n1b, othx, othxb);
    { float* tf = curx; curx = othx; othx = tf;
      u16*  tb = curxb; curxb = othxb; othxb = tb; }

    // MoE (dense over all experts, masked by top-2 softmax weights)
    gate_kernel<<<(T * E + 255) / 256, 256, 0, stream>>>(curx, gw, gb, gs);
    topk_kernel<<<(T + 255) / 256, 256, 0, stream>>>(gs, we);
    zero_kernel<<<(unsigned)(((size_t)T * D + 255) / 256), 256, 0, stream>>>(macc, (size_t)T * D);
    for (int e = 0; e < E; e++){
      gemm_kernel<<<dim3(F / 128, T / 128), 256, 0, stream>>>(
          curxb, D, w1t + (size_t)e * F * D, D, b1 + (size_t)e * F,
          nullptr, 0, hf, F, hb, D, 1, 0);
      gemm_kernel<<<dim3(D / 128, T / 128), 256, 0, stream>>>(
          hb, F, w2t + (size_t)e * D * F, F, b2 + (size_t)e * D,
          we + e, E, macc, D, nullptr, F, 0, 1);
    }
    add_ln_kernel<<<T, 256, 0, stream>>>(curx, macc, n2g, n2b, othx, othxb);
    { float* tf = curx; curx = othx; othx = tf;
      u16*  tb = curxb; curxb = othxb; othxb = tb; }
  }

  // final LN + head projection straight into d_out
  add_ln_kernel<<<T, 256, 0, stream>>>(curx, nullptr, lnf_g, lnf_b, othx, othxb);
  gemm_kernel<<<dim3(V / 128, T / 128), 256, 0, stream>>>(
      othxb, D, hwb, D, head_b, nullptr, 0, (float*)d_out, V, nullptr, D, 0, 0);
}
